// GATEncoder_40381282517238
// MI455X (gfx1250) — matrix operand order repaired and checked by reading, not compiled, not run on hardware
//
#include <hip/hip_runtime.h>
#include <hip/hip_bf16.h>

// ---------- types for WMMA ----------
typedef __attribute__((ext_vector_type(16))) __bf16 v16bf;
typedef __attribute__((ext_vector_type(8)))  float  v8f;

#define NN    10000
#define EE    160000
#define ET    170000   // EE + NN self loops
#define NB    40       // ceil(NN/256) scan blocks

__device__ inline unsigned short f32_to_bf16_rne(float f) {
    unsigned u = __float_as_uint(f);
    unsigned r = 0x7FFFu + ((u >> 16) & 1u);
    return (unsigned short)((u + r) >> 16);
}

// ---------- generic fill ----------
__global__ void k_fill_u32(unsigned* p, unsigned v, size_t n) {
    size_t i = (size_t)blockIdx.x * blockDim.x + threadIdx.x;
    if (i < n) p[i] = v;
}

// ---------- edge list with self loops ----------
__global__ void k_build_edges(const int* __restrict__ ei, int* __restrict__ src,
                              int* __restrict__ dst, int E, int n) {
    int i = blockIdx.x * blockDim.x + threadIdx.x;
    int Et = E + n;
    if (i >= Et) return;
    if (i < E) { src[i] = ei[i]; dst[i] = ei[E + i]; }
    else       { src[i] = i - E; dst[i] = i - E; }
}

// ---------- CSR construction ----------
__global__ void k_count(const int* __restrict__ dst, int* __restrict__ cnt, int Et) {
    int i = blockIdx.x * blockDim.x + threadIdx.x;
    if (i < Et) atomicAdd(&cnt[dst[i]], 1);
}
__global__ void k_scan_block(const int* __restrict__ in, int* __restrict__ out,
                             int* __restrict__ bsum, int n) {
    __shared__ int sh[256];
    int i = blockIdx.x * 256 + threadIdx.x;
    int v = (i < n) ? in[i] : 0;
    sh[threadIdx.x] = v;
    __syncthreads();
    for (int o = 1; o < 256; o <<= 1) {
        int t = (threadIdx.x >= o) ? sh[threadIdx.x - o] : 0;
        __syncthreads();
        sh[threadIdx.x] += t;
        __syncthreads();
    }
    if (i < n) out[i] = sh[threadIdx.x] - v;  // exclusive
    if (threadIdx.x == 255) bsum[blockIdx.x] = sh[255];
}
__global__ void k_scan_sums(int* bsum, int nb) {
    if (threadIdx.x == 0 && blockIdx.x == 0) {
        int acc = 0;
        for (int b = 0; b < nb; ++b) { int v = bsum[b]; bsum[b] = acc; acc += v; }
    }
}
__global__ void k_scan_add(int* __restrict__ out, const int* __restrict__ bsum, int n) {
    int i = blockIdx.x * 256 + threadIdx.x;
    if (i < n) out[i] += bsum[blockIdx.x];
}
__global__ void k_scatter(const int* __restrict__ dst, int* __restrict__ cursor,
                          int* __restrict__ eord, int Et) {
    int i = blockIdx.x * blockDim.x + threadIdx.x;
    if (i < Et) { int p = atomicAdd(&cursor[dst[i]], 1); eord[p] = i; }
}

// ---------- fp32 -> bf16 ----------
__global__ void k_conv_bf16(const float* __restrict__ x, unsigned short* __restrict__ y, size_t n) {
    size_t i = (size_t)blockIdx.x * blockDim.x + threadIdx.x;
    if (i < n) y[i] = f32_to_bf16_rne(x[i]);
}

// ---------- fp32 [K,N] -> bf16 transposed [N,K] ----------
__global__ void k_conv_w_t(const float* __restrict__ W, unsigned short* __restrict__ Wt,
                           int K, int Nc) {
    size_t i = (size_t)blockIdx.x * blockDim.x + threadIdx.x;
    if (i >= (size_t)K * Nc) return;
    int k = (int)(i / Nc);
    int nc = (int)(i - (size_t)k * Nc);
    Wt[(size_t)nc * K + k] = f32_to_bf16_rne(W[i]);
}

// ---------- WMMA fragments ----------
struct Frag { union { v16bf v; unsigned u[8]; }; };

__device__ inline void ld_frag_a(Frag& f, const unsigned* p, int kh) {
#pragma unroll
    for (int i = 0; i < 4; ++i) f.u[i]     = p[kh * 4 + i];
#pragma unroll
    for (int i = 0; i < 4; ++i) f.u[4 + i] = p[8 + kh * 4 + i];
}
__device__ inline void ld_frag_b(Frag& f, const unsigned* p, int kh) {
#pragma unroll
    for (int j = 0; j < 4; ++j) f.u[j]     = p[kh * 4 + j];
#pragma unroll
    for (int j = 0; j < 4; ++j) f.u[4 + j] = p[8 + kh * 4 + j];
}

// ---------- WMMA GEMM: C[M,Nc] = A[M,K] @ Bt[Nc,K]^T (+bias) ----------
// Block = 8 waves; tile M=128, N=64. Each wave: 16-row strip, 4 accumulators.
// 2-stage software pipeline over K (K % 64 == 0 for all calls).
__global__ __launch_bounds__(256) void k_wmma_gemm(
    const unsigned short* __restrict__ A, const unsigned short* __restrict__ Bt,
    const float* __restrict__ bias, float* __restrict__ C, int M, int K, int Nc) {
    const int wave = threadIdx.x >> 5;
    const int lane = threadIdx.x & 31;
    const int m0 = blockIdx.y * 128 + wave * 16;
    if (m0 >= M) return;                       // wave-uniform: EXEC stays all-1s
    const int n0 = blockIdx.x * 64;
    const int lr = lane & 15;
    const int kh = lane >> 4;
    const int mrow = m0 + lr;

    v8f acc[4];
#pragma unroll
    for (int t = 0; t < 4; ++t) {
        float b = bias ? bias[n0 + t * 16 + lr] : 0.0f;
#pragma unroll
        for (int r = 0; r < 8; ++r) acc[t][r] = b;
    }

    const unsigned* ar = (const unsigned*)(A + (size_t)mrow * K);
    const unsigned* br[4];
#pragma unroll
    for (int t = 0; t < 4; ++t)
        br[t] = (const unsigned*)(Bt + (size_t)(n0 + t * 16 + lr) * K);

    Frag a0, a1, b0[4], b1[4];
    ld_frag_a(a0, ar, kh);
#pragma unroll
    for (int t = 0; t < 4; ++t) ld_frag_b(b0[t], br[t], kh);

    for (int k0 = 0; k0 < K; k0 += 64) {
        const int kd = k0 >> 1;  // dword offset of current stage
        // stage 2 loads (k0+32)
        ld_frag_a(a1, ar + kd + 16, kh);
#pragma unroll
        for (int t = 0; t < 4; ++t) ld_frag_b(b1[t], br[t] + kd + 16, kh);
        __builtin_prefetch(ar + kd + 96, 0, 1);  // A streams from HBM
#pragma unroll
        for (int t = 0; t < 4; ++t)
            acc[t] = __builtin_amdgcn_wmma_f32_16x16x32_bf16(
                false, a0.v, false, b0[t].v, (short)0, acc[t], false, false);
        if (k0 + 64 < K) {
            ld_frag_a(a0, ar + kd + 32, kh);
#pragma unroll
            for (int t = 0; t < 4; ++t) ld_frag_b(b0[t], br[t] + kd + 32, kh);
        }
#pragma unroll
        for (int t = 0; t < 4; ++t)
            acc[t] = __builtin_amdgcn_wmma_f32_16x16x32_bf16(
                false, a1.v, false, b1[t].v, (short)0, acc[t], false, false);
    }

    // C/D layout: VGPR r -> M = r + 8*kh, N = lr
#pragma unroll
    for (int t = 0; t < 4; ++t) {
#pragma unroll
        for (int r = 0; r < 8; ++r) {
            C[(size_t)(m0 + r + 8 * kh) * Nc + n0 + t * 16 + lr] = acc[t][r];
        }
    }
}

// ---------- attention dots: a_src[n,h] = <H[n,h,:], att_src[h,:]>, same for dst ----------
__global__ void k_attn_dots(const float* __restrict__ Hf, const float* __restrict__ att_s,
                            const float* __restrict__ att_d, float* __restrict__ a_s,
                            float* __restrict__ a_d, int n, int Hh, int Cc) {
    int wid  = (int)(((size_t)blockIdx.x * blockDim.x + threadIdx.x) >> 5);
    int lane = threadIdx.x & 31;
    if (wid >= n * Hh) return;
    int node = wid / Hh, h = wid - node * Hh;
    const float* hv = Hf + (size_t)node * Hh * Cc + (size_t)h * Cc;
    float s1 = 0.f, s2 = 0.f;
    for (int c = lane; c < Cc; c += 32) {
        float x = hv[c];
        s1 += x * att_s[h * Cc + c];
        s2 += x * att_d[h * Cc + c];
    }
    for (int o = 16; o > 0; o >>= 1) {
        s1 += __shfl_down(s1, o);
        s2 += __shfl_down(s2, o);
    }
    if (lane == 0) { a_s[wid] = s1; a_d[wid] = s2; }
}

// ---------- per-dst softmax over incoming edges (one wave per node, no atomics) ----------
__global__ void k_softmax_csr(const int* __restrict__ rowptr, const int* __restrict__ eord,
                              const int* __restrict__ src, const float* __restrict__ a_s,
                              const float* __restrict__ a_d, float* __restrict__ ebuf,
                              float* __restrict__ invden, int n, int Hh) {
    int wid  = (int)(((size_t)blockIdx.x * blockDim.x + threadIdx.x) >> 5);
    int lane = threadIdx.x & 31;
    if (wid >= n) return;
    int beg = rowptr[wid], end = rowptr[wid + 1];
    for (int h = 0; h < Hh; ++h) {
        float ad = a_d[wid * Hh + h];
        float mx = -3.4e38f;
        for (int p = beg + lane; p < end; p += 32) {
            int eid = eord[p];
            float v = a_s[src[eid] * Hh + h] + ad;
            v = (v > 0.f) ? v : 0.2f * v;
            mx = fmaxf(mx, v);
        }
        for (int o = 16; o > 0; o >>= 1) mx = fmaxf(mx, __shfl_xor(mx, o));
        float sm = 0.f;
        for (int p = beg + lane; p < end; p += 32) {
            int eid = eord[p];
            float v = a_s[src[eid] * Hh + h] + ad;
            v = (v > 0.f) ? v : 0.2f * v;
            float w = __expf(v - mx);
            ebuf[(size_t)eid * Hh + h] = w;
            sm += w;
        }
        for (int o = 16; o > 0; o >>= 1) sm += __shfl_xor(sm, o);
        if (lane == 0) invden[wid * Hh + h] = 1.0f / sm;
    }
}

// ---------- per-dst aggregation + fused bias (+ELU): one block per node, no atomics ----------
__global__ __launch_bounds__(256) void k_agg_csr(
    const int* __restrict__ rowptr, const int* __restrict__ eord, const int* __restrict__ src,
    const float* __restrict__ ebuf, const float* __restrict__ invden,
    const float* __restrict__ Hf, const float* __restrict__ bias, float* __restrict__ out,
    int Hh, int Cc, int do_elu) {
    int d  = blockIdx.x;
    int t  = threadIdx.x;
    int HC = Hh * Cc;
    int nch = HC >> 8;  // HC / 256 (2048->8, 512->2)
    int beg = rowptr[d], end = rowptr[d + 1];

    float inv[4];
    for (int h = 0; h < Hh; ++h) inv[h] = invden[d * Hh + h];
    int hd[8];
#pragma unroll
    for (int i = 0; i < 8; ++i) hd[i] = (i < nch) ? ((t + 256 * i) / Cc) : 0;

    float acc[8];
#pragma unroll
    for (int i = 0; i < 8; ++i) acc[i] = 0.f;

    for (int p = beg; p < end; ++p) {
        int eid = eord[p];
        int s   = src[eid];
        const float* hs = Hf + (size_t)s * HC;
        for (int i = 0; i < nch; ++i) {
            float w = ebuf[(size_t)eid * Hh + hd[i]] * inv[hd[i]];
            acc[i] += w * hs[t + 256 * i];
        }
    }
    float* od = out + (size_t)d * HC;
    for (int i = 0; i < nch; ++i) {
        int c = t + 256 * i;
        float v = acc[i] + bias[c];
        if (do_elu) v = (v > 0.f) ? v : (__expf(v) - 1.0f);
        od[c] = v;
    }
}

extern "C" void kernel_launch(void* const* d_in, const int* in_sizes, int n_in,
                              void* d_out, int out_size, void* d_ws, size_t ws_size,
                              hipStream_t stream) {
    (void)in_sizes; (void)n_in; (void)out_size; (void)ws_size;
    const float* x      = (const float*)d_in[0];
    const int*   ei     = (const int*)d_in[1];
    const float* proj_W = (const float*)d_in[2];
    const float* proj_b = (const float*)d_in[3];
    const float* W1  = (const float*)d_in[4];
    const float* as1 = (const float*)d_in[5];
    const float* ad1 = (const float*)d_in[6];
    const float* b1  = (const float*)d_in[7];
    const float* W2  = (const float*)d_in[8];
    const float* as2 = (const float*)d_in[9];
    const float* ad2 = (const float*)d_in[10];
    const float* b2  = (const float*)d_in[11];
    const float* W3  = (const float*)d_in[12];
    const float* as3 = (const float*)d_in[13];
    const float* ad3 = (const float*)d_in[14];
    const float* b3  = (const float*)d_in[15];

    // ---- workspace carve-up ----
    char* ws = (char*)d_ws;
    size_t off = 0;
    auto take = [&](size_t bytes) -> void* {
        void* p = ws + off;
        off = (off + bytes + 255) & ~(size_t)255;
        return p;
    };
    unsigned short* XB = (unsigned short*)take((size_t)NN * 2048 * 2);   // bf16 features
    unsigned short* WT = (unsigned short*)take((size_t)2048 * 2048 * 2); // bf16 W^T
    float* F0 = (float*)take((size_t)NN * 2048 * 4);  // current features (fp32)
    float* Hf = (float*)take((size_t)NN * 2048 * 4);  // GEMM output h
    float* a_s = (float*)take((size_t)NN * 4 * 4);
    float* a_d = (float*)take((size_t)NN * 4 * 4);
    float* invden = (float*)take((size_t)NN * 4 * 4);
    float* ebuf = (float*)take((size_t)ET * 4 * 4);
    int* srcA   = (int*)take((size_t)ET * 4);
    int* dstA   = (int*)take((size_t)ET * 4);
    int* rowptr = (int*)take((size_t)(NN + 1) * 4);
    int* cnt    = (int*)take((size_t)NN * 4);
    int* cursor = (int*)take((size_t)NN * 4);
    int* bsum   = (int*)take((size_t)64 * 4);
    int* eord   = (int*)take((size_t)ET * 4);

    auto blks = [](size_t n, int t) { return (unsigned)((n + t - 1) / t); };

    // ---- edges with self loops + CSR-by-destination ----
    k_build_edges<<<blks(ET, 256), 256, 0, stream>>>(ei, srcA, dstA, EE, NN);
    k_fill_u32<<<blks(NN, 256), 256, 0, stream>>>((unsigned*)cnt, 0u, NN);
    k_count<<<blks(ET, 256), 256, 0, stream>>>(dstA, cnt, ET);
    k_scan_block<<<NB, 256, 0, stream>>>(cnt, rowptr, bsum, NN);
    k_scan_sums<<<1, 32, 0, stream>>>(bsum, NB);
    k_scan_add<<<NB, 256, 0, stream>>>(rowptr, bsum, NN);
    k_fill_u32<<<1, 32, 0, stream>>>((unsigned*)(rowptr + NN), (unsigned)ET, 1);
    hipMemcpyAsync(cursor, rowptr, (size_t)NN * 4, hipMemcpyDeviceToDevice, stream);
    k_scatter<<<blks(ET, 256), 256, 0, stream>>>(dstA, cursor, eord, ET);

    // ---- projection: F0 = x @ proj_W + proj_b  (N x 512) ----
    k_conv_bf16<<<blks((size_t)NN * 256, 256), 256, 0, stream>>>(x, XB, (size_t)NN * 256);
    k_conv_w_t<<<blks((size_t)256 * 512, 256), 256, 0, stream>>>(proj_W, WT, 256, 512);
    {
        dim3 g(512 / 64, (NN + 127) / 128);
        k_wmma_gemm<<<g, 256, 0, stream>>>(XB, WT, proj_b, F0, NN, 256, 512);
    }

    // ---- one GAT layer ----
    auto gat_layer = [&](const float* W, const float* as, const float* ad, const float* bl,
                         int K, int Hh, int Cc, float* dstbuf, int do_elu) {
        int HC = Hh * Cc;
        k_conv_bf16<<<blks((size_t)NN * K, 256), 256, 0, stream>>>(F0, XB, (size_t)NN * K);
        k_conv_w_t<<<blks((size_t)K * HC, 256), 256, 0, stream>>>(W, WT, K, HC);
        dim3 g(HC / 64, (NN + 127) / 128);
        k_wmma_gemm<<<g, 256, 0, stream>>>(XB, WT, nullptr, Hf, NN, K, HC);

        k_attn_dots<<<blks((size_t)NN * Hh * 32, 256), 256, 0, stream>>>(Hf, as, ad, a_s, a_d,
                                                                         NN, Hh, Cc);
        k_softmax_csr<<<blks((size_t)NN * 32, 256), 256, 0, stream>>>(rowptr, eord, srcA, a_s,
                                                                      a_d, ebuf, invden, NN, Hh);
        k_agg_csr<<<NN, 256, 0, stream>>>(rowptr, eord, srcA, ebuf, invden, Hf, bl, dstbuf,
                                          Hh, Cc, do_elu);
    };

    gat_layer(W1, as1, ad1, b1, 512, 4, 512, F0, 1);             // -> F0 [N,2048]
    gat_layer(W2, as2, ad2, b2, 2048, 4, 512, F0, 1);            // -> F0 [N,2048]
    gat_layer(W3, as3, ad3, b3, 2048, 1, 512, (float*)d_out, 0); // -> out [N,512]
}
